// MemoryEnhancedMoE_38242388803632
// MI455X (gfx1250) — matrix operand order repaired and checked by reading, not compiled, not run on hardware
//
#include <hip/hip_runtime.h>
#include <math.h>

typedef __attribute__((ext_vector_type(16))) __bf16 v16bf;
typedef __attribute__((ext_vector_type(8)))  float  v8f;
typedef __attribute__((ext_vector_type(4)))  float  f32x4;

union Frag { v16bf v; f32x4 q[2]; };

#define WMMA_BF16(a, b, c) \
  __builtin_amdgcn_wmma_f32_16x16x32_bf16(false, (a), false, (b), (short)0, (c), false, false)

static constexpr int B_ROWS     = 4096;
static constexpr int IN_DIM     = 1024;
static constexpr int N_EXPERTS  = 16;
static constexpr int GATE_HID   = 256;
static constexpr int EMB        = 512;
static constexpr int N_MEM      = 32768;
static constexpr int TOP_K      = 5;
static constexpr int CHUNK      = 256;              // sim columns per chunk
static constexpr int NCHUNK     = N_MEM / CHUNK;    // 128
static constexpr int OUT_STRIDE = N_EXPERTS + TOP_K + IN_DIM; // 1045

// ---------------------------------------------------------------- helpers
__device__ __forceinline__ void ins5(float v, int ix, float bv[5], int bi[5]) {
  if (v <= bv[4]) return;
  bv[4] = v; bi[4] = ix;
#pragma unroll
  for (int j = 4; j > 0; --j) {
    if (bv[j] > bv[j - 1]) {
      float tv = bv[j]; bv[j] = bv[j - 1]; bv[j - 1] = tv;
      int   ti = bi[j]; bi[j] = bi[j - 1]; bi[j - 1] = ti;
    }
  }
}

// ---------------------------------------------------------------- fp32 -> bf16
__global__ void k_tobf16(const float* __restrict__ in, __bf16* __restrict__ out, int n) {
  int i = blockIdx.x * 256 + threadIdx.x;
  if (i < n) out[i] = (__bf16)in[i];
}

// W[K][N] fp32 -> Wt[N][K] bf16
__global__ void k_transpose(const float* __restrict__ W, __bf16* __restrict__ Wt,
                            int K, int N) {
  int i = blockIdx.x * 256 + threadIdx.x;
  if (i < K * N) {
    int n = i / K, k = i - n * K;
    Wt[i] = (__bf16)W[(size_t)k * N + n];
  }
}

// ------------------------------------------------- gating hidden: relu(x@gW1+gb1)
__global__ __launch_bounds__(256) void k_gate1(const __bf16* __restrict__ X,
                                               const __bf16* __restrict__ W1t,
                                               const float* __restrict__ b1,
                                               float* __restrict__ GH) {
  __shared__ __bf16 sA[16][1032];
  const int tid = threadIdx.x, lane = tid & 31, wave = tid >> 5;
  const int lr = lane & 15, hi = lane >> 4;
  const size_t rowBase = (size_t)blockIdx.x * 16;

  for (int i = tid; i < 16 * 128; i += 256) {
    int r = i >> 7, c = (i & 127) * 8;
    *(f32x4*)&sA[r][c] = *(const f32x4*)&X[(rowBase + r) * IN_DIM + c];
  }
  __syncthreads();

  v8f zero = {0, 0, 0, 0, 0, 0, 0, 0};
  v8f acc[2] = {zero, zero};
  const int nb = wave * 32;
  for (int kt = 0; kt < 32; ++kt) {
    const int kk = kt * 32;
    Frag fa;
    const __bf16* pa = &sA[lr][kk + hi * 16];
    fa.q[0] = *(const f32x4*)pa;
    fa.q[1] = *(const f32x4*)(pa + 8);
#pragma unroll
    for (int a = 0; a < 2; ++a) {
      Frag fb;
      const __bf16* pb = &W1t[(size_t)(nb + a * 16 + lr) * IN_DIM + kk + hi * 16];
      fb.q[0] = *(const f32x4*)pb;
      fb.q[1] = *(const f32x4*)(pb + 8);
      acc[a] = WMMA_BF16(fa.v, fb.v, acc[a]);
    }
  }
#pragma unroll
  for (int a = 0; a < 2; ++a)
#pragma unroll
    for (int j = 0; j < 8; ++j) {
      int r = j + hi * 8, c = nb + a * 16 + lr;
      GH[(rowBase + r) * GATE_HID + c] = fmaxf(acc[a][j] + b1[c], 0.f);
    }
}

// ------------------------------------------------- gating logits + softmax -> out
__global__ __launch_bounds__(256) void k_gate2(const float* __restrict__ GH,
                                               const float* __restrict__ W2,
                                               const float* __restrict__ b2,
                                               float* __restrict__ out) {
  __shared__ float sL[16][16];
  const int tid = threadIdx.x, r = tid >> 4, e = tid & 15;
  const size_t row = (size_t)blockIdx.x * 16 + r;
  float acc = b2[e];
  const float* gh = &GH[row * GATE_HID];
  for (int k = 0; k < GATE_HID; ++k) acc += gh[k] * W2[k * N_EXPERTS + e];
  sL[r][e] = acc;
  __syncthreads();
  float mx = sL[r][0];
  for (int i = 1; i < 16; ++i) mx = fmaxf(mx, sL[r][i]);
  float sum = 0.f;
  for (int i = 0; i < 16; ++i) sum += expf(sL[r][i] - mx);
  out[row * OUT_STRIDE + e] = expf(acc - mx) / sum;
}

// ------------------------------------------------- encoder layer 1: LN(Z@W1+b1), relu
__global__ __launch_bounds__(256) void k_enc1(const __bf16* __restrict__ Z,
                                              const __bf16* __restrict__ W1t,
                                              const float* __restrict__ b1,
                                              const float* __restrict__ lg,
                                              const float* __restrict__ lb,
                                              __bf16* __restrict__ H) {
  __shared__ __align__(16) char smem[16 * 1032 * 2]; // reused: A tile then score tile
  __bf16(*sA)[1032] = reinterpret_cast<__bf16(*)[1032]>(smem);
  float(*sS)[516]   = reinterpret_cast<float(*)[516]>(smem);
  __shared__ float redS[16][16], redQ[16][16], stats[16][2];

  const int tid = threadIdx.x, lane = tid & 31, wave = tid >> 5;
  const int lr = lane & 15, hi = lane >> 4;
  const size_t rowBase = (size_t)blockIdx.x * 16;

  for (int i = tid; i < 16 * 128; i += 256) {
    int r = i >> 7, c = (i & 127) * 8;
    *(f32x4*)&sA[r][c] = *(const f32x4*)&Z[(rowBase + r) * IN_DIM + c];
  }
  __syncthreads();

  v8f zero = {0, 0, 0, 0, 0, 0, 0, 0};
  v8f acc[4] = {zero, zero, zero, zero};
  const int nb = wave * 64;
  for (int kt = 0; kt < 32; ++kt) {
    const int kk = kt * 32;
    Frag fa;
    const __bf16* pa = &sA[lr][kk + hi * 16];
    fa.q[0] = *(const f32x4*)pa;
    fa.q[1] = *(const f32x4*)(pa + 8);
#pragma unroll
    for (int a = 0; a < 4; ++a) {
      Frag fb;
      const __bf16* pb = &W1t[(size_t)(nb + a * 16 + lr) * IN_DIM + kk + hi * 16];
      fb.q[0] = *(const f32x4*)pb;
      fb.q[1] = *(const f32x4*)(pb + 8);
      acc[a] = WMMA_BF16(fa.v, fb.v, acc[a]);
    }
  }
  __syncthreads(); // done reading sA; reuse as score tile
#pragma unroll
  for (int a = 0; a < 4; ++a)
#pragma unroll
    for (int j = 0; j < 8; ++j) {
      int r = j + hi * 8, c = nb + a * 16 + lr;
      sS[r][c] = acc[a][j] + b1[c];
    }
  __syncthreads();

  const int row = tid >> 4, seg = tid & 15;
  float s = 0.f, s2 = 0.f;
  for (int c = seg * 32; c < seg * 32 + 32; ++c) {
    float v = sS[row][c]; s += v; s2 += v * v;
  }
  redS[row][seg] = s; redQ[row][seg] = s2;
  __syncthreads();
  if (seg == 0) {
    float S = 0.f, Q = 0.f;
    for (int i = 0; i < 16; ++i) { S += redS[row][i]; Q += redQ[row][i]; }
    float mu = S * (1.f / EMB);
    float var = Q * (1.f / EMB) - mu * mu;
    stats[row][0] = mu;
    stats[row][1] = rsqrtf(var + 1e-5f);
  }
  __syncthreads();
  float mu = stats[row][0], inv = stats[row][1];
  for (int c = seg * 32; c < seg * 32 + 32; ++c) {
    float v = (sS[row][c] - mu) * inv * lg[c] + lb[c];
    H[(rowBase + row) * EMB + c] = (__bf16)fmaxf(v, 0.f);
  }
}

// ------------------------- encoder layer 2: LN(H@W2+b2), then L2-normalize row
__global__ __launch_bounds__(256) void k_enc2(const __bf16* __restrict__ H,
                                              const __bf16* __restrict__ W2t,
                                              const float* __restrict__ b2,
                                              const float* __restrict__ lg,
                                              const float* __restrict__ lb,
                                              __bf16* __restrict__ En) {
  __shared__ __bf16 sA[16][520];
  __shared__ float  sS[16][516];
  __shared__ float  redS[16][16], redQ[16][16], stats[16][2];

  const int tid = threadIdx.x, lane = tid & 31, wave = tid >> 5;
  const int lr = lane & 15, hi = lane >> 4;
  const size_t rowBase = (size_t)blockIdx.x * 16;

  for (int i = tid; i < 16 * 64; i += 256) {
    int r = i >> 6, c = (i & 63) * 8;
    *(f32x4*)&sA[r][c] = *(const f32x4*)&H[(rowBase + r) * EMB + c];
  }
  __syncthreads();

  v8f zero = {0, 0, 0, 0, 0, 0, 0, 0};
  v8f acc[4] = {zero, zero, zero, zero};
  const int nb = wave * 64;
  for (int kt = 0; kt < 16; ++kt) {
    const int kk = kt * 32;
    Frag fa;
    const __bf16* pa = &sA[lr][kk + hi * 16];
    fa.q[0] = *(const f32x4*)pa;
    fa.q[1] = *(const f32x4*)(pa + 8);
#pragma unroll
    for (int a = 0; a < 4; ++a) {
      Frag fb;
      const __bf16* pb = &W2t[(size_t)(nb + a * 16 + lr) * EMB + kk + hi * 16];
      fb.q[0] = *(const f32x4*)pb;
      fb.q[1] = *(const f32x4*)(pb + 8);
      acc[a] = WMMA_BF16(fa.v, fb.v, acc[a]);
    }
  }
#pragma unroll
  for (int a = 0; a < 4; ++a)
#pragma unroll
    for (int j = 0; j < 8; ++j) {
      int r = j + hi * 8, c = nb + a * 16 + lr;
      sS[r][c] = acc[a][j] + b2[c];
    }
  __syncthreads();

  const int row = tid >> 4, seg = tid & 15;
  float s = 0.f, s2 = 0.f;
  for (int c = seg * 32; c < seg * 32 + 32; ++c) {
    float v = sS[row][c]; s += v; s2 += v * v;
  }
  redS[row][seg] = s; redQ[row][seg] = s2;
  __syncthreads();
  if (seg == 0) {
    float S = 0.f, Q = 0.f;
    for (int i = 0; i < 16; ++i) { S += redS[row][i]; Q += redQ[row][i]; }
    float mu = S * (1.f / EMB);
    float var = Q * (1.f / EMB) - mu * mu;
    stats[row][0] = mu;
    stats[row][1] = rsqrtf(var + 1e-5f);
  }
  __syncthreads();
  float mu = stats[row][0], inv = stats[row][1];
  float q2 = 0.f;
  for (int c = seg * 32; c < seg * 32 + 32; ++c) {
    float y = (sS[row][c] - mu) * inv * lg[c] + lb[c];
    sS[row][c] = y;
    q2 += y * y;
  }
  redS[row][seg] = q2;
  __syncthreads();
  if (seg == 0) {
    float Q = 0.f;
    for (int i = 0; i < 16; ++i) Q += redS[row][i];
    stats[row][0] = 1.f / (sqrtf(Q) + 1e-8f);
  }
  __syncthreads();
  float invn = stats[row][0];
  for (int c = seg * 32; c < seg * 32 + 32; ++c)
    En[(rowBase + row) * EMB + c] = (__bf16)(sS[row][c] * invn);
}

// ----------------- fused sims GEMM (qn @ mn^T) + per-chunk top-5 reduction
__global__ __launch_bounds__(256) void k_sims(const __bf16* __restrict__ Q,
                                              const __bf16* __restrict__ Mn,
                                              float* __restrict__ cV,
                                              int* __restrict__ cI) {
  __shared__ __bf16 sA[16][520];
  __shared__ float  sS[16][260];
  __shared__ float  cv[16][16][TOP_K];
  __shared__ int    ci[16][16][TOP_K];

  const int tid = threadIdx.x, lane = tid & 31, wave = tid >> 5;
  const int lr = lane & 15, hi = lane >> 4;
  const size_t rowBase = (size_t)blockIdx.x * 16;
  const int row = tid >> 4, seg = tid & 15;

  for (int i = tid; i < 16 * 64; i += 256) {
    int r = i >> 6, c = (i & 63) * 8;
    *(f32x4*)&sA[r][c] = *(const f32x4*)&Q[(rowBase + r) * EMB + c];
  }
  __syncthreads();

  v8f zero = {0, 0, 0, 0, 0, 0, 0, 0};
  for (int ch = 0; ch < NCHUNK; ++ch) {
    const int n0 = ch * CHUNK;
    v8f acc0 = zero, acc1 = zero;
    for (int kt = 0; kt < 16; ++kt) {
      const int kk = kt * 32;
      Frag fa;
      const __bf16* pa = &sA[lr][kk + hi * 16];
      fa.q[0] = *(const f32x4*)pa;
      fa.q[1] = *(const f32x4*)(pa + 8);
      Frag fb0, fb1;
      const __bf16* pb0 = &Mn[(size_t)(n0 + wave * 32 + lr) * EMB + kk + hi * 16];
      fb0.q[0] = *(const f32x4*)pb0;
      fb0.q[1] = *(const f32x4*)(pb0 + 8);
      acc0 = WMMA_BF16(fa.v, fb0.v, acc0);
      const __bf16* pb1 = &Mn[(size_t)(n0 + wave * 32 + 16 + lr) * EMB + kk + hi * 16];
      fb1.q[0] = *(const f32x4*)pb1;
      fb1.q[1] = *(const f32x4*)(pb1 + 8);
      acc1 = WMMA_BF16(fa.v, fb1.v, acc1);
    }
    __syncthreads(); // prior chunk consumers of sS/cv are done
#pragma unroll
    for (int j = 0; j < 8; ++j) {
      int r = j + hi * 8;
      sS[r][wave * 32 + lr]      = acc0[j];
      sS[r][wave * 32 + 16 + lr] = acc1[j];
    }
    __syncthreads();
    float bv[5] = {-1e30f, -1e30f, -1e30f, -1e30f, -1e30f};
    int   bi[5] = {-1, -1, -1, -1, -1};
    for (int c = 0; c < 16; ++c) {
      int cc = seg * 16 + c;
      ins5(sS[row][cc], n0 + cc, bv, bi);
    }
#pragma unroll
    for (int j = 0; j < 5; ++j) { cv[row][seg][j] = bv[j]; ci[row][seg][j] = bi[j]; }
    __syncthreads();
    if (seg == 0) {
      float mv[5] = {-1e30f, -1e30f, -1e30f, -1e30f, -1e30f};
      int   mi[5] = {-1, -1, -1, -1, -1};
      for (int s = 0; s < 16; ++s)
#pragma unroll
        for (int j = 0; j < 5; ++j) ins5(cv[row][s][j], ci[row][s][j], mv, mi);
      size_t base = (rowBase + row) * (size_t)(NCHUNK * TOP_K) + (size_t)ch * TOP_K;
#pragma unroll
      for (int j = 0; j < 5; ++j) { cV[base + j] = mv[j]; cI[base + j] = mi[j]; }
    }
  }
}

// ----------------- merge 640 chunk candidates per row -> global top-5
__global__ void k_merge(const float* __restrict__ cV, const int* __restrict__ cI,
                        float* __restrict__ topv, int* __restrict__ topi) {
  int row = blockIdx.x * 256 + threadIdx.x;
  if (row >= B_ROWS) return;
  float bv[5] = {-1e30f, -1e30f, -1e30f, -1e30f, -1e30f};
  int   bi[5] = {-1, -1, -1, -1, -1};
  const float* v = &cV[(size_t)row * NCHUNK * TOP_K];
  const int*  ix = &cI[(size_t)row * NCHUNK * TOP_K];
  for (int t = 0; t < NCHUNK * TOP_K; ++t) ins5(v[t], ix[t], bv, bi);
#pragma unroll
  for (int j = 0; j < 5; ++j) { topv[row * 5 + j] = bv[j]; topi[row * 5 + j] = bi[j]; }
}

// ----------------- gather + similarity-weighted mean -> out[, 16:1045]
__global__ __launch_bounds__(256) void k_combine(const float* __restrict__ topv,
                                                 const int* __restrict__ topi,
                                                 const float* __restrict__ contents,
                                                 float* __restrict__ out) {
  __shared__ float w[5];
  __shared__ int   ix[5];
  const int tid = threadIdx.x;
  const size_t row = blockIdx.x;
  if (tid < 5) {
    float v = topv[row * 5 + tid];
    float wv = (v >= 0.f) ? v : 0.f;
    w[tid] = wv;
    ix[tid] = topi[row * 5 + tid];
    out[row * OUT_STRIDE + N_EXPERTS + tid] = wv;
  }
  __syncthreads();
  float inv = 1.f / (w[0] + w[1] + w[2] + w[3] + w[4] + 1e-8f);
  const int cc = tid * 4;
  f32x4 accv = {0, 0, 0, 0};
#pragma unroll
  for (int j = 0; j < 5; ++j) {
    f32x4 cvv = *(const f32x4*)&contents[(size_t)ix[j] * IN_DIM + cc];
    accv += cvv * w[j];
  }
  float* o = &out[row * OUT_STRIDE + N_EXPERTS + TOP_K + cc];
  o[0] = accv[0] * inv; o[1] = accv[1] * inv;
  o[2] = accv[2] * inv; o[3] = accv[3] * inv;
}

// ================================================================ launcher
extern "C" void kernel_launch(void* const* d_in, const int* in_sizes, int n_in,
                              void* d_out, int out_size, void* d_ws, size_t ws_size,
                              hipStream_t stream) {
  (void)in_sizes; (void)n_in; (void)out_size; (void)ws_size;
  const float* x    = (const float*)d_in[0];
  const float* cont = (const float*)d_in[1];
  const float* gW1  = (const float*)d_in[2];
  const float* gb1  = (const float*)d_in[3];
  const float* gW2  = (const float*)d_in[4];
  const float* gb2  = (const float*)d_in[5];
  const float* eW1  = (const float*)d_in[6];
  const float* eb1  = (const float*)d_in[7];
  const float* ln1g = (const float*)d_in[8];
  const float* ln1b = (const float*)d_in[9];
  const float* eW2  = (const float*)d_in[10];
  const float* eb2  = (const float*)d_in[11];
  const float* ln2g = (const float*)d_in[12];
  const float* ln2b = (const float*)d_in[13];
  float* out = (float*)d_out;

  char* ws = (char*)d_ws;
  size_t off = 0;
  auto take = [&](size_t bytes) -> char* {
    char* p = ws + off;
    off = (off + bytes + 255) & ~(size_t)255;
    return p;
  };
  __bf16* xb    = (__bf16*)take((size_t)B_ROWS * IN_DIM * 2);
  __bf16* cb    = (__bf16*)take((size_t)N_MEM * IN_DIM * 2);
  __bf16* eW1t  = (__bf16*)take((size_t)EMB * IN_DIM * 2);
  __bf16* eW2t  = (__bf16*)take((size_t)EMB * EMB * 2);
  __bf16* gW1t  = (__bf16*)take((size_t)GATE_HID * IN_DIM * 2);
  __bf16* Hx    = (__bf16*)take((size_t)B_ROWS * EMB * 2);
  __bf16* Hc    = (__bf16*)take((size_t)N_MEM * EMB * 2);
  __bf16* qn    = (__bf16*)take((size_t)B_ROWS * EMB * 2);
  __bf16* mn    = (__bf16*)take((size_t)N_MEM * EMB * 2);
  float*  ghid  = (float*)take((size_t)B_ROWS * GATE_HID * 4);
  float*  cV    = (float*)take((size_t)B_ROWS * NCHUNK * TOP_K * 4);
  int*    cI    = (int*)take((size_t)B_ROWS * NCHUNK * TOP_K * 4);
  float*  topv  = (float*)take((size_t)B_ROWS * TOP_K * 4);
  int*    topi  = (int*)take((size_t)B_ROWS * TOP_K * 4);

  // 1) bf16 conversions + weight transposes
  k_tobf16<<<(B_ROWS * IN_DIM + 255) / 256, 256, 0, stream>>>(x, xb, B_ROWS * IN_DIM);
  k_tobf16<<<(N_MEM * IN_DIM + 255) / 256, 256, 0, stream>>>(cont, cb, N_MEM * IN_DIM);
  k_transpose<<<(IN_DIM * GATE_HID + 255) / 256, 256, 0, stream>>>(gW1, gW1t, IN_DIM, GATE_HID);
  k_transpose<<<(IN_DIM * EMB + 255) / 256, 256, 0, stream>>>(eW1, eW1t, IN_DIM, EMB);
  k_transpose<<<(EMB * EMB + 255) / 256, 256, 0, stream>>>(eW2, eW2t, EMB, EMB);

  // 2) gating path
  k_gate1<<<B_ROWS / 16, 256, 0, stream>>>(xb, gW1t, gb1, ghid);
  k_gate2<<<B_ROWS / 16, 256, 0, stream>>>(ghid, gW2, gb2, out);

  // 3) encoder (queries + memory bank)
  k_enc1<<<B_ROWS / 16, 256, 0, stream>>>(xb, eW1t, eb1, ln1g, ln1b, Hx);
  k_enc1<<<N_MEM / 16, 256, 0, stream>>>(cb, eW1t, eb1, ln1g, ln1b, Hc);
  k_enc2<<<B_ROWS / 16, 256, 0, stream>>>(Hx, eW2t, eb2, ln2g, ln2b, qn);
  k_enc2<<<N_MEM / 16, 256, 0, stream>>>(Hc, eW2t, eb2, ln2g, ln2b, mn);

  // 4) fused similarity GEMM + chunked top-k, then merge + combine
  k_sims<<<B_ROWS / 16, 256, 0, stream>>>(qn, mn, cV, cI);
  k_merge<<<B_ROWS / 256, 256, 0, stream>>>(cV, cI, topv, topi);
  k_combine<<<B_ROWS, 256, 0, stream>>>(topv, topi, cont, out);
}
